// DoubleSelfAttention_89146341195873
// MI455X (gfx1250) — compile-verified, matrix operand order checked by reading, not run on hardware
//
#include <hip/hip_runtime.h>

typedef __attribute__((ext_vector_type(16))) _Float16 v16h;
typedef __attribute__((ext_vector_type(8)))  _Float16 v8h;
typedef __attribute__((ext_vector_type(8)))  float    v8f;
typedef __attribute__((ext_vector_type(4))) unsigned int u32x4;
typedef __attribute__((ext_vector_type(8))) int          i32x8;
typedef __attribute__((ext_vector_type(4))) int          i32x4;

#define B_   4
#define C_   256
#define H1_  32
#define W1_  32
#define H2_  64
#define W2_  64
#define N_   4096
#define EPS_ 1e-5f
#define SLOPE_ 0.1f
#define VROW 40   // padded LDS row stride in halves (80B: conflict-free b128 tile reads)

#if defined(__AMDGCN__) && __has_builtin(__builtin_amdgcn_tensor_load_to_lds)
#define HAVE_TDM 1
#else
#define HAVE_TDM 0
#endif

#if defined(__AMDGCN__) && __has_builtin(__builtin_amdgcn_permlane16)
#define HAVE_PERMLANE 1
#else
#define HAVE_PERMLANE 0
#endif

// Load a 16x32 f16 tile (row-major, row stride ldRow halves) into WMMA A/B
// operand layout (ISA 7.12.2): lane L holds row (L&15); lanes 0-15 carry
// K={0..7,16..23}, lanes 16-31 carry K={8..15,24..31}.
__device__ __forceinline__ v16h ld_tile16(const _Float16* p, int ldRow) {
  const int lane = threadIdx.x & 31;
  const int r    = lane & 15;
  const int ks   = (lane & 16) ? 8 : 0;
  const _Float16* q = p + r * ldRow + ks;
  v8h lo = *(const v8h*)(q);
  v8h hi = *(const v8h*)(q + 16);
  v16h out;
#pragma unroll
  for (int i = 0; i < 8; ++i) { out[i] = lo[i]; out[i + 8] = hi[i]; }
  return out;
}

// max-reduce over each 16-lane half (all lanes receive the result).
// VALU path via v_permlane16_b32 (xor patterns), DS fallback via __shfl_xor.
__device__ __forceinline__ float xmax16(float x) {
#if HAVE_PERMLANE
  unsigned v, t;
  v = __float_as_uint(x);
  t = __builtin_amdgcn_permlane16(v, v, 0x67452301u, 0xEFCDAB89u, false, false); // xor 1
  x = fmaxf(x, __uint_as_float(t)); v = __float_as_uint(x);
  t = __builtin_amdgcn_permlane16(v, v, 0x54761032u, 0xDCFE98BAu, false, false); // xor 2
  x = fmaxf(x, __uint_as_float(t)); v = __float_as_uint(x);
  t = __builtin_amdgcn_permlane16(v, v, 0x32107654u, 0xBA98FEDCu, false, false); // xor 4
  x = fmaxf(x, __uint_as_float(t)); v = __float_as_uint(x);
  t = __builtin_amdgcn_permlane16(v, v, 0xFEDCBA98u, 0x76543210u, false, false); // xor 8
  x = fmaxf(x, __uint_as_float(t));
#else
  x = fmaxf(x, __shfl_xor(x, 1, 32));
  x = fmaxf(x, __shfl_xor(x, 2, 32));
  x = fmaxf(x, __shfl_xor(x, 4, 32));
  x = fmaxf(x, __shfl_xor(x, 8, 32));
#endif
  return x;
}

#if HAVE_TDM
// TDM: async copy of a [256 rows x 32 halves] tile (global stride N_ halves)
// into LDS at ldsOff, inserting 16B padding after every 64B row (VROW=40).
__device__ __forceinline__ void tdm_stage_v(const _Float16* gsrc, unsigned ldsOff) {
  unsigned long long ga = (unsigned long long)(size_t)gsrc;
  u32x4 g0 = { 1u,                                   // count=1 (valid), user mode
               ldsOff,                               // lds_addr (bytes)
               (unsigned)ga,                         // global_addr[31:0]
               (unsigned)(ga >> 32) | (2u << 30) };  // global_addr[56:32] | type=2
  i32x8 g1 = { (int)0x06D10000,   // data_size=2B | pad_en | interval=64B | amount=16B
               (int)0x10000000,   // tensor_dim0 = 4096 (low16 at bits 63:48)
               (int)0x01000000,   // tensor_dim1 = 256  (low16 at bits 95:80)
               (int)0x00200000,   // tile_dim0 = 32     (bits 127:112)
               (int)0x00000100,   // tile_dim1 = 256, tile_dim2 = 0
               (int)0x00001000,   // tensor_dim0_stride = 4096
               0, 0 };            // tensor_dim1_stride = 0 (2D)
  i32x4 gz4 = { 0, 0, 0, 0 };
#if __clang_major__ >= 23
  i32x8 gz8 = { 0, 0, 0, 0, 0, 0, 0, 0 };
  __builtin_amdgcn_tensor_load_to_lds(g0, g1, gz4, gz4, gz8, 0);
#else
  __builtin_amdgcn_tensor_load_to_lds(g0, g1, gz4, gz4, 0);
#endif
}
#endif

// ---------------- bilinear upsample x1 [B,C,32,32] -> x1u [B,C,64,64] -------
__global__ void upsample_bilinear(const float* __restrict__ x1,
                                  float* __restrict__ x1u) {
  int idx = blockIdx.x * blockDim.x + threadIdx.x;
  if (idx >= B_ * C_ * H2_ * W2_) return;
  int w2 = idx & (W2_ - 1);
  int h2 = (idx >> 6) & (H2_ - 1);
  int bc = idx >> 12;
  float fy = fminf(fmaxf((h2 + 0.5f) * 0.5f - 0.5f, 0.f), (float)(H1_ - 1));
  float fx = fminf(fmaxf((w2 + 0.5f) * 0.5f - 0.5f, 0.f), (float)(W1_ - 1));
  int y0 = min((int)fy, H1_ - 2);
  int x0 = min((int)fx, W1_ - 2);
  float wy = fy - y0, wx = fx - x0;
  const float* src = x1 + bc * (H1_ * W1_);
  float v00 = src[y0 * W1_ + x0],     v01 = src[y0 * W1_ + x0 + 1];
  float v10 = src[(y0+1) * W1_ + x0], v11 = src[(y0+1) * W1_ + x0 + 1];
  float top = v00 + wx * (v01 - v00);
  float bot = v10 + wx * (v11 - v10);
  x1u[idx] = top + wy * (bot - top);
}

// ------------- q/k projection: [32,C]x[C,N] + BN + LReLU -> f16 [B,N,32] ----
__global__ void proj32_kernel(const float* __restrict__ x,   // [B,C,N]
                              const float* __restrict__ w,   // [32,C]
                              const float* __restrict__ g,  const float* __restrict__ bb,
                              const float* __restrict__ rm, const float* __restrict__ rv,
                              _Float16* __restrict__ outT) { // [B,N,32]
  const int lane = threadIdx.x & 31;
  const int wid  = threadIdx.x >> 5;
  const int nblk = N_ / 8;
  const int b = blockIdx.x / nblk;
  const int n = (blockIdx.x % nblk) * 8 + wid;
  const int d = lane;
  const float* xb = x + (size_t)b * C_ * N_ + n;
  const float* wr = w + d * C_;
  float acc = 0.f;
#pragma unroll 4
  for (int c = 0; c < C_; ++c) acc += wr[c] * xb[(size_t)c * N_];
  float sc = g[d] * rsqrtf(rv[d] + EPS_);
  float y  = acc * sc + (bb[d] - rm[d] * sc);
  y = (y >= 0.f) ? y : SLOPE_ * y;
  outT[((size_t)b * N_ + n) * 32 + d] = (_Float16)y;
}

// ------------- v projection: [C,C]x[C,N] + BN + LReLU -> f16 [B,C,N] --------
// 8 output channels per lane per pass -> 8 FMAs per LDS load.
__global__ void projv_kernel(const float* __restrict__ x2,  // [B,C,N]
                             const float* __restrict__ wv,  // [C,C]
                             const float* __restrict__ g,  const float* __restrict__ bb,
                             const float* __restrict__ rm, const float* __restrict__ rv,
                             _Float16* __restrict__ vh) {   // [B,C,N]
  __shared__ float xs[C_ * 32];
  const int lane = threadIdx.x & 31;
  const int wid  = threadIdx.x >> 5;
  const int nblk = N_ / 32;
  const int b  = blockIdx.x / nblk;
  const int n0 = (blockIdx.x % nblk) * 32;
  const float* xb = x2 + (size_t)b * C_ * N_;
  for (int i = threadIdx.x; i < C_ * 32; i += 256) {
    int cc = i >> 5, nn = i & 31;
    xs[i] = xb[(size_t)cc * N_ + n0 + nn];
  }
  __syncthreads();
  const int c0 = wid * 32;
#pragma unroll
  for (int grp = 0; grp < 4; ++grp) {
    const int cb = c0 + grp * 8;
    float acc[8];
#pragma unroll
    for (int u = 0; u < 8; ++u) acc[u] = 0.f;
    for (int cc = 0; cc < C_; ++cc) {
      float xv = xs[cc * 32 + lane];
#pragma unroll
      for (int u = 0; u < 8; ++u) acc[u] += wv[(size_t)(cb + u) * C_ + cc] * xv;
    }
#pragma unroll
    for (int u = 0; u < 8; ++u) {
      int c = cb + u;
      float sc = g[c] * rsqrtf(rv[c] + EPS_);
      float y  = acc[u] * sc + (bb[c] - rm[c] * sc);
      y = (y >= 0.f) ? y : SLOPE_ * y;
      vh[((size_t)b * C_ + c) * N_ + n0 + lane] = (_Float16)y;
    }
  }
}

// ------------- fused flash attention + gamma*out + x1u ----------------------
// 256 threads = 8 waves; wave w owns query rows m0..m0+15 (block = 128 rows).
// Per 32-key chunk: V chunk staged in LDS by the TDM (double-buffered,
// prefetching next chunk), 2 energy WMMAs, online softmax (permlane16 max,
// row-sums via an all-ones WMMA so the sum lands in the out layout),
// P transposed via per-wave LDS tile, 16 accumulate WMMAs from LDS V tiles.
__global__ void __launch_bounds__(256)
flash_attn_kernel(const _Float16* __restrict__ qT, const _Float16* __restrict__ kT,
                  const _Float16* __restrict__ vh, const float* __restrict__ x1u,
                  const float* __restrict__ gamma, float* __restrict__ out) {
  __shared__ __align__(16) _Float16 Vs[2][C_ * VROW];   // 2 x 20KB (padded rows)
  __shared__ __align__(16) _Float16 Pbuf[8][16 * VROW];
  __shared__ float abuf[8][16];

  const int lane = threadIdx.x & 31;
  const int wid  = threadIdx.x >> 5;
  const int half = lane >> 4;
  const int nl   = lane & 15;

  const int b    = blockIdx.x >> 5;      // 32 m-blocks per batch
  const int mblk = blockIdx.x & 31;
  const int m0   = mblk * 128 + wid * 16;

  const _Float16* qTb = qT + (size_t)b * N_ * 32;
  const _Float16* kTb = kT + (size_t)b * N_ * 32;
  const _Float16* vb  = vh + (size_t)b * C_ * N_;
  const float*    xub = x1u + (size_t)b * C_ * N_;
  float*          ob  = out + (size_t)b * C_ * N_;

  const v16h a_q = ld_tile16(qTb + (size_t)m0 * 32, 32);  // A = q^T tile, K=32

  v16h a_one;
#pragma unroll
  for (int i = 0; i < 16; ++i) a_one[i] = (_Float16)1.0f;

  v8f zinit = {};
  v8f acc[16];
#pragma unroll
  for (int t = 0; t < 16; ++t) acc[t] = zinit;
  v8f lsum = zinit;                       // row-sums of P, kept in out layout
  float mrun[8];
#pragma unroll
  for (int j = 0; j < 8; ++j) mrun[j] = -1e30f;

#if HAVE_TDM
  const unsigned vsOff0 = (unsigned)(size_t)&Vs[0][0];
  const unsigned vsOff1 = (unsigned)(size_t)&Vs[1][0];
  if (wid == 0) {
    tdm_stage_v(vb, vsOff0);                       // stage chunk 0
    __builtin_amdgcn_s_wait_tensorcnt(0);
  }
  __syncthreads();
#endif
  int buf = 0;

  for (int n0 = 0; n0 < N_; n0 += 32) {
#if HAVE_TDM
    if (wid == 0 && n0 + 32 < N_)                  // async prefetch next chunk
      tdm_stage_v(vb + (n0 + 32), (buf == 0) ? vsOff1 : vsOff0);
#else
    __syncthreads();                               // reads of previous chunk done
    {
      int cid = threadIdx.x;
#pragma unroll
      for (int kk = 0; kk < 4; ++kk) {
        int c = cid >> 2, qq = cid & 3;
        *(v8h*)&Vs[0][c * VROW + qq * 8] =
            *(const v8h*)(vb + (size_t)c * N_ + n0 + qq * 8);
        cid += 256;
      }
    }
    __syncthreads();
#endif

    v16h b_k0 = ld_tile16(kTb + (size_t)n0 * 32, 32);
    v16h b_k1 = ld_tile16(kTb + (size_t)(n0 + 16) * 32, 32);
    v8f zero = {};
    v8f e0 = __builtin_amdgcn_wmma_f32_16x16x32_f16(false, a_q, false, b_k0,
                                                    (short)0, zero, false, false);
    v8f e1 = __builtin_amdgcn_wmma_f32_16x16x32_f16(false, a_q, false, b_k1,
                                                    (short)0, zero, false, false);

    // D layout: lane holds col n=nl, rows m=j+8*half -> row max over 16 lanes.
    float p0[8], p1[8], alpha[8];
#pragma unroll
    for (int j = 0; j < 8; ++j) {
      float r = xmax16(fmaxf(e0[j], e1[j]));
      float mnew = fmaxf(mrun[j], r);
      alpha[j] = __expf(mrun[j] - mnew);
      p0[j] = __expf(e0[j] - mnew);
      p1[j] = __expf(e1[j] - mnew);
      mrun[j] = mnew;
    }
    if (nl == 0) {
#pragma unroll
      for (int j = 0; j < 8; ++j) abuf[wid][half * 8 + j] = alpha[j];
    }
#pragma unroll
    for (int j = 0; j < 8; ++j) {        // store P as [m][n] f16, per-wave
      int m = half * 8 + j;
      Pbuf[wid][m * VROW + nl]      = (_Float16)p0[j];
      Pbuf[wid][m * VROW + 16 + nl] = (_Float16)p1[j];
    }
    // per-wave LDS comms: same-wave DS ordering is in-order; fence the compiler
    asm volatile("" ::: "memory");

    float acol = abuf[wid][nl];          // out-column rescale factor
    v16h b_p = ld_tile16(&Pbuf[wid][0], VROW);  // B = P^T (rows=n, cols=m)

    {                                    // row-sums of P via ones-WMMA
      v8f ct = lsum;
#pragma unroll
      for (int j = 0; j < 8; ++j) ct[j] *= acol;
      lsum = __builtin_amdgcn_wmma_f32_16x16x32_f16(false, a_one, false, b_p,
                                                    (short)0, ct, false, false);
    }

    const _Float16* vsb = &Vs[buf][0];
#pragma unroll
    for (int t = 0; t < 16; ++t) {
      v8f ct = acc[t];
#pragma unroll
      for (int j = 0; j < 8; ++j) ct[j] *= acol;
      v16h a_v = ld_tile16(vsb + t * 16 * VROW, VROW);
      acc[t] = __builtin_amdgcn_wmma_f32_16x16x32_f16(false, a_v, false, b_p,
                                                      (short)0, ct, false, false);
    }

#if HAVE_TDM
    if (wid == 0) __builtin_amdgcn_s_wait_tensorcnt(0);
    __syncthreads();                     // publish next chunk, retire this one
    buf ^= 1;
#endif
  }

  // lsum's D rows are all identical (sum over n of P[m][n]) -> no transpose.
  float linv = 1.0f / lsum[0];
  float gm = gamma[0];
  int m = m0 + nl;
#pragma unroll
  for (int t = 0; t < 16; ++t) {
#pragma unroll
    for (int j = 0; j < 8; ++j) {
      int c = t * 16 + half * 8 + j;
      size_t off = (size_t)c * N_ + m;
      ob[off] = gm * (acc[t][j] * linv) + xub[off];
    }
  }
}

extern "C" void kernel_launch(void* const* d_in, const int* in_sizes, int n_in,
                              void* d_out, int out_size, void* d_ws, size_t ws_size,
                              hipStream_t stream) {
  const float* x1 = (const float*)d_in[0];
  const float* x2 = (const float*)d_in[1];
  const float* wq = (const float*)d_in[2];
  const float* wk = (const float*)d_in[3];
  const float* wv = (const float*)d_in[4];
  const float* gq = (const float*)d_in[5];
  const float* bq = (const float*)d_in[6];
  const float* mq = (const float*)d_in[7];
  const float* vq = (const float*)d_in[8];
  const float* gk = (const float*)d_in[9];
  const float* bk = (const float*)d_in[10];
  const float* mk = (const float*)d_in[11];
  const float* vk = (const float*)d_in[12];
  const float* gv = (const float*)d_in[13];
  const float* bv = (const float*)d_in[14];
  const float* mv = (const float*)d_in[15];
  const float* vv = (const float*)d_in[16];
  const float* gamma = (const float*)d_in[17];
  float* out = (float*)d_out;

  char* ws = (char*)d_ws;
  float*    x1u = (float*)ws;                                 // 16,777,216 B
  _Float16* qT  = (_Float16*)(ws + 16777216);                 //  1,048,576 B
  _Float16* kT  = (_Float16*)(ws + 16777216 + 1048576);       //  1,048,576 B
  _Float16* vh  = (_Float16*)(ws + 16777216 + 2097152);       //  8,388,608 B

  upsample_bilinear<<<(B_ * C_ * H2_ * W2_) / 256, 256, 0, stream>>>(x1, x1u);
  proj32_kernel<<<B_ * (N_ / 8), 256, 0, stream>>>(x1u, wq, gq, bq, mq, vq, qT);
  proj32_kernel<<<B_ * (N_ / 8), 256, 0, stream>>>(x2,  wk, gk, bk, mk, vk, kT);
  projv_kernel<<<B_ * (N_ / 32), 256, 0, stream>>>(x2, wv, gv, bv, mv, vv, vh);
  flash_attn_kernel<<<B_ * (N_ / 128), 256, 0, stream>>>(qT, kT, vh, x1u, gamma, out);
}
